// ResidualBlock_20194936226236
// MI455X (gfx1250) — matrix operand order repaired and checked by reading, not compiled, not run on hardware
//
#include <hip/hip_runtime.h>
#include <hip/hip_bf16.h>

// ---------------- problem constants (from reference) ----------------
#define Nn      100000
#define Ee      1600000
#define INd     128
#define Hh      4
#define Cc      32
#define HC      128
#define EDd     16
#define ETOT    (Ee + Nn)          // 1,700,000 edges incl. self loops
#define NEG_SLOPE 0.2f
#define EPS_LN  1e-5f

#define MT      (Nn / 16)          // 6250 row tiles (exact)
#define TMG     ((MT + 7) / 8)     // m-tile groups of 8 (one per block)

// ---------------- workspace layout (in floats) ----------------------
// zero-initialized region first (one contiguous memset-like kernel)
#define OFF_EASUM  ((size_t)0)                       // N*ED
#define OFF_CNT    (OFF_EASUM + (size_t)Nn*EDd)      // N
#define OFF_AMAX   (OFF_CNT   + (size_t)Nn)          // N*H (u32 encoded)
#define OFF_DENOM  (OFF_AMAX  + (size_t)Nn*Hh)       // N*H
#define OFF_OUTACC (OFF_DENOM + (size_t)Nn*Hh)       // N*HC
#define OFF_STATS  (OFF_OUTACC+ (size_t)Nn*HC)       // 8 (sum, sumsq)
#define ZERO_END   (OFF_STATS + (size_t)8)
// non-zeroed scratch
#define OFF_SELF   (ZERO_END)                        // N*ED
#define OFF_XL     (OFF_SELF  + (size_t)Nn*EDd)      // N*HC
#define OFF_XR     (OFF_XL    + (size_t)Nn*HC)       // N*HC
#define OFF_ALPHA  (OFF_XR    + (size_t)Nn*HC)       // ETOT*H (logits)

typedef __attribute__((ext_vector_type(16))) __bf16 v16bf;
typedef __attribute__((ext_vector_type(8)))  float  v8f;

// monotonic float<->u32 encoding for atomic max over signed floats
__device__ __forceinline__ unsigned fenc(float f) {
    unsigned u = __float_as_uint(f);
    return (u & 0x80000000u) ? ~u : (u | 0x80000000u);
}
__device__ __forceinline__ float fdec(unsigned u) {
    u = (u & 0x80000000u) ? (u & 0x7fffffffu) : ~u;
    return __uint_as_float(u);
}

// -------------------- K1: zero the accumulator region ----------------
__global__ void k_zero(float* ws) {
    size_t i = (size_t)blockIdx.x * blockDim.x + threadIdx.x;
    if (i < ZERO_END) ws[i] = 0.0f;
}

// -------------------- K2: edge_attr segment sum / count --------------
__global__ void k_ea_scatter(const int* __restrict__ ei,
                             const float* __restrict__ ea,
                             float* __restrict__ ea_sum,
                             float* __restrict__ cnt) {
    int e = blockIdx.x * blockDim.x + threadIdx.x;
    if (e >= Ee) return;
    int d = ei[Ee + e];
    atomicAdd(&cnt[d], 1.0f);
    const float* src = ea + (size_t)e * EDd;
    float* dst = ea_sum + (size_t)d * EDd;
#pragma unroll
    for (int k = 0; k < EDd; ++k) atomicAdd(&dst[k], src[k]);
}

// -------------------- K3: self-loop attr = mean ----------------------
__global__ void k_self_attr(const float* __restrict__ ea_sum,
                            const float* __restrict__ cnt,
                            float* __restrict__ self_attr) {
    size_t i = (size_t)blockIdx.x * blockDim.x + threadIdx.x;
    if (i >= (size_t)Nn * EDd) return;
    float c = cnt[i / EDd];
    c = (c < 1.0f) ? 1.0f : c;
    self_attr[i] = ea_sum[i] / c;
}

// -------------------- K4: WMMA GEMM  xl = x@Wl+bl, xr = x@Wr+br ------
// block = 8 waves sharing one (matrix, col-tile); the 128x16 weight
// panel is staged once into LDS as bf16 (col-major, padded) and each
// wave reads B fragments with two ds_load_b128 per k-step.
#define WPAD 136   // 128 + 8 element pad -> conflict-free ds_load_b128
__global__ void k_gemm_xlxr(const float* __restrict__ x,
                            const float* __restrict__ Wl, const float* __restrict__ bl,
                            const float* __restrict__ Wr, const float* __restrict__ br,
                            float* __restrict__ xl, float* __restrict__ xr) {
    __shared__ __bf16 lds_w[16 * WPAD];          // [col][k] bf16 panel

    int bid = blockIdx.x;
    const int g     = bid % TMG;  bid /= TMG;
    const int tn    = bid & 7;
    const int which = bid >> 3;

    const float* W    = which ? Wr : Wl;
    const float* bias = which ? br : bl;
    float*       out  = which ? xr : xl;

    // cooperative stage: 128K x 16cols, read row-major (coalesced),
    // store transposed col-major into LDS as bf16
#pragma unroll
    for (int r = 0; r < 8; ++r) {
        int idx = threadIdx.x + r * 256;         // 2048 elements
        int k = idx >> 4, c = idx & 15;
        lds_w[c * WPAD + k] = (__bf16)W[(size_t)k * HC + tn * 16 + c];
    }
    __syncthreads();

    const int waveId = threadIdx.x >> 5;
    const int lane   = threadIdx.x & 31;
    const int tm     = g * 8 + waveId;
    if (tm >= MT) return;

    const int hi   = lane >> 4;
    const int lmod = lane & 15;
    const int arow = tm * 16 + lmod;             // A row owned by this lane

    v8f acc = {};
#pragma unroll
    for (int kb = 0; kb < INd; kb += 32) {
        // ---- A tile 16x32 bf16: lane-lo K kb..kb+15,
        //      lane-hi K kb+16..kb+31
        const float* ap = x + (size_t)arow * INd + kb + (hi ? 16 : 0);
        v16bf a;
#pragma unroll
        for (int i = 0; i < 8; ++i) a[i]     = (__bf16)ap[i];
#pragma unroll
        for (int i = 0; i < 8; ++i) a[8 + i] = (__bf16)ap[8 + i];
        // ---- B tile 32x16 bf16 from LDS: lane-lo K kb..kb+15,
        //      lane-hi K kb+16..kb+31, column = lane&15 (contiguous 32B)
        const v16bf b = *(const v16bf*)(lds_w + lmod * WPAD + kb + (hi ? 16 : 0));

        acc = __builtin_amdgcn_wmma_f32_16x16x32_bf16(
            /*neg_a=*/false, a, /*neg_b=*/false, b,
            /*c_mod=*/(short)0, acc, /*reuse_a=*/false, /*reuse_b=*/false);
    }

    const float bv = bias[tn * 16 + lmod];
    // C layout: VGPR i -> row (i + 8*hi), column = lane&15
#pragma unroll
    for (int i = 0; i < 8; ++i) {
        int m = tm * 16 + i + (hi ? 8 : 0);
        out[(size_t)m * HC + tn * 16 + lmod] = acc[i] + bv;
    }
}

// -------------------- K5: attention logits + segment max -------------
// one wave per edge; lane l owns channels 4l..4l+3 (head = l>>3).
// We (8KB) and att staged in LDS per block (serves 8 edge-waves).
__global__ void k_attn(const int* __restrict__ ei,
                       const float* __restrict__ edge_attr,
                       const float* __restrict__ self_attr,
                       const float* __restrict__ xl,
                       const float* __restrict__ xr,
                       const float* __restrict__ We,
                       const float* __restrict__ att,
                       float* __restrict__ alpha,
                       unsigned* __restrict__ amax) {
    __shared__ float s_We[EDd * HC];   // 8KB
    __shared__ float s_att[HC];        // 512B
#pragma unroll
    for (int r = 0; r < 8; ++r) s_We[threadIdx.x + r * 256] = We[threadIdx.x + r * 256];
    if (threadIdx.x < HC) s_att[threadIdx.x] = att[threadIdx.x];
    __syncthreads();

    const int wave = (int)((blockIdx.x * blockDim.x + threadIdx.x) >> 5);
    const int lane = threadIdx.x & 31;
    if (wave >= ETOT) return;
    const int e = wave;
    int s, d; const float* ea;
    if (e < Ee) { s = ei[e]; d = ei[Ee + e]; ea = edge_attr + (size_t)e * EDd; }
    else        { s = d = e - Ee;            ea = self_attr + (size_t)(e - Ee) * EDd; }

    const float4 xlv = *(const float4*)(xl + (size_t)s * HC + 4 * lane);
    const float4 xrv = *(const float4*)(xr + (size_t)d * HC + 4 * lane);

    float eav[EDd];
#pragma unroll
    for (int k = 0; k < EDd; ++k) eav[k] = ea[k];     // broadcast reads

    float4 em = {0.f, 0.f, 0.f, 0.f};                  // (e @ We), 4 channels
#pragma unroll
    for (int k = 0; k < EDd; ++k) {
        const float4 w = *(const float4*)(s_We + (size_t)k * HC + 4 * lane);
        em.x = fmaf(eav[k], w.x, em.x);
        em.y = fmaf(eav[k], w.y, em.y);
        em.z = fmaf(eav[k], w.z, em.z);
        em.w = fmaf(eav[k], w.w, em.w);
    }
    float m0 = xlv.x + xrv.x + em.x;
    float m1 = xlv.y + xrv.y + em.y;
    float m2 = xlv.z + xrv.z + em.z;
    float m3 = xlv.w + xrv.w + em.w;
    m0 = (m0 > 0.f) ? m0 : NEG_SLOPE * m0;
    m1 = (m1 > 0.f) ? m1 : NEG_SLOPE * m1;
    m2 = (m2 > 0.f) ? m2 : NEG_SLOPE * m2;
    m3 = (m3 > 0.f) ? m3 : NEG_SLOPE * m3;

    const float4 av = *(const float4*)(s_att + 4 * lane);
    float p = m0 * av.x + m1 * av.y + m2 * av.z + m3 * av.w;
    // reduce the 8 lanes of this head
    p += __shfl_xor(p, 1, 32);
    p += __shfl_xor(p, 2, 32);
    p += __shfl_xor(p, 4, 32);
    if ((lane & 7) == 0) {
        const int h = lane >> 3;
        alpha[(size_t)e * Hh + h] = p;                 // store logit
        atomicMax(&amax[(size_t)d * Hh + h], fenc(p));
    }
}

// -------------------- K6: fused exp + denom + scatter-aggregate ------
// aggregate with UNNORMALIZED exp weights; per-node normalization is
// deferred to the residual pass (saves a whole edge pass).
__global__ void k_aggregate(const int* __restrict__ ei,
                            const float* __restrict__ xl,
                            const float* __restrict__ alpha,
                            const unsigned* __restrict__ amax,
                            float* __restrict__ denom,
                            float* __restrict__ outacc) {
    const int wave = (int)((blockIdx.x * blockDim.x + threadIdx.x) >> 5);
    const int lane = threadIdx.x & 31;
    if (wave >= ETOT) return;
    const int e = wave;
    int s, d;
    if (e < Ee) { s = ei[e]; d = ei[Ee + e]; }
    else        { s = d = e - Ee; }
    const int h = lane >> 3;
    const float logit = alpha[(size_t)e * Hh + h];         // broadcast in 8-lane group
    const float mx    = fdec(amax[(size_t)d * Hh + h]);
    const float v     = __expf(logit - mx);
    if ((lane & 7) == 0) atomicAdd(&denom[(size_t)d * Hh + h], v);

    const float4 xv = *(const float4*)(xl + (size_t)s * HC + 4 * lane);
    float* o = outacc + (size_t)d * HC + 4 * lane;
    atomicAdd(o + 0, xv.x * v);
    atomicAdd(o + 1, xv.y * v);
    atomicAdd(o + 2, xv.z * v);
    atomicAdd(o + 3, xv.w * v);
}

// -------------------- K7: residual (with deferred /denom) + stats ----
__global__ void k_resid_stats(const float* __restrict__ x,
                              const float* __restrict__ outacc,
                              const float* __restrict__ denom,
                              const float* __restrict__ bias,
                              float* __restrict__ z,
                              float* __restrict__ stats) {
    __shared__ float s1[256], s2[256];
    size_t t = (size_t)blockIdx.x * blockDim.x + threadIdx.x;
    float zv = 0.0f;
    if (t < (size_t)Nn * HC) {
        const int c = (int)(t % HC);
        const size_t n = t / HC;
        const float dnm = denom[n * Hh + (c >> 5)];
        zv = x[t] + outacc[t] / dnm + bias[c];
        z[t] = zv;
    }
    s1[threadIdx.x] = zv;
    s2[threadIdx.x] = zv * zv;
    __syncthreads();
    for (int off = 128; off > 0; off >>= 1) {
        if (threadIdx.x < off) {
            s1[threadIdx.x] += s1[threadIdx.x + off];
            s2[threadIdx.x] += s2[threadIdx.x + off];
        }
        __syncthreads();
    }
    if (threadIdx.x == 0) {
        atomicAdd(&stats[0], s1[0]);
        atomicAdd(&stats[1], s2[0]);
    }
}

// -------------------- K8: layernorm + relu ---------------------------
__global__ void k_norm_relu(float* __restrict__ z,
                            const float* __restrict__ lnw,
                            const float* __restrict__ lnb,
                            const float* __restrict__ stats) {
    size_t t = (size_t)blockIdx.x * blockDim.x + threadIdx.x;
    if (t >= (size_t)Nn * HC) return;
    const float inv = 1.0f / (float)((size_t)Nn * HC);
    const float mu  = stats[0] * inv;
    const float var = stats[1] * inv - mu * mu;
    const float r   = rsqrtf(var + EPS_LN);
    const int   c   = (int)(t % HC);
    float v = (z[t] - mu) * r * lnw[c] + lnb[c];
    z[t] = (v > 0.f) ? v : 0.f;
}

// ====================================================================
extern "C" void kernel_launch(void* const* d_in, const int* in_sizes, int n_in,
                              void* d_out, int out_size, void* d_ws, size_t ws_size,
                              hipStream_t stream) {
    const float* x    = (const float*)d_in[0];
    const int*   ei   = (const int*)  d_in[1];     // [2,E]: src row then dst row
    const float* ea   = (const float*)d_in[2];
    const float* Wl   = (const float*)d_in[3];
    const float* bl   = (const float*)d_in[4];
    const float* Wr   = (const float*)d_in[5];
    const float* br   = (const float*)d_in[6];
    const float* We   = (const float*)d_in[7];
    const float* att  = (const float*)d_in[8];
    const float* bias = (const float*)d_in[9];
    const float* lnw  = (const float*)d_in[10];
    const float* lnb  = (const float*)d_in[11];
    float* out = (float*)d_out;

    float*    ws      = (float*)d_ws;
    float*    ea_sum  = ws + OFF_EASUM;
    float*    cnt     = ws + OFF_CNT;
    unsigned* amax    = (unsigned*)(ws + OFF_AMAX);
    float*    denom   = ws + OFF_DENOM;
    float*    outacc  = ws + OFF_OUTACC;
    float*    stats   = ws + OFF_STATS;
    float*    selfat  = ws + OFF_SELF;
    float*    xl      = ws + OFF_XL;
    float*    xr      = ws + OFF_XR;
    float*    alpha   = ws + OFF_ALPHA;

    const int TB = 256;
    // K1: zero accumulators
    {
        int blocks = (int)((ZERO_END + TB - 1) / TB);
        k_zero<<<blocks, TB, 0, stream>>>(ws);
    }
    // K2: edge_attr segment sum / count over dst
    k_ea_scatter<<<(Ee + TB - 1) / TB, TB, 0, stream>>>(ei, ea, ea_sum, cnt);
    // K3: self-loop attr
    {
        size_t n = (size_t)Nn * EDd;
        k_self_attr<<<(int)((n + TB - 1) / TB), TB, 0, stream>>>(ea_sum, cnt, selfat);
    }
    // K4: WMMA node transforms; block = (which, tn, 8 m-tiles)
    {
        int blocks = 2 * 8 * TMG;
        k_gemm_xlxr<<<blocks, TB, 0, stream>>>(x, Wl, bl, Wr, br, xl, xr);
    }
    // K5: attention logits + segment max (one wave per edge; ETOT % 8 == 0)
    {
        long long threads = (long long)ETOT * 32;
        k_attn<<<(int)((threads + TB - 1) / TB), TB, 0, stream>>>(
            ei, ea, selfat, xl, xr, We, att, alpha, amax);
    }
    // K6: fused exp + denominator + aggregation scatter
    {
        long long threads = (long long)ETOT * 32;
        k_aggregate<<<(int)((threads + TB - 1) / TB), TB, 0, stream>>>(
            ei, xl, alpha, amax, denom, outacc);
    }
    // K7: residual + deferred normalization + whole-tensor stats
    {
        size_t n = (size_t)Nn * HC;
        k_resid_stats<<<(int)((n + TB - 1) / TB), TB, 0, stream>>>(
            x, outacc, denom, bias, out, stats);
    }
    // K8: layernorm + relu
    {
        size_t n = (size_t)Nn * HC;
        k_norm_relu<<<(int)((n + TB - 1) / TB), TB, 0, stream>>>(out, lnw, lnb, stats);
    }
}
